// UniCoreLinearFP4_8504035246563
// MI455X (gfx1250) — matrix operand order repaired and checked by reading, not compiled, not run on hardware
//
#include <hip/hip_runtime.h>

// ---------------------------------------------------------------------------
// Fused dequant GEMM:  out = x @ (w * s)^T + bias
//   x: [M,K] f32, w: [N,K] f32, s: [N] f32, bias: [N] f32, out: [M,N] f32
// 3xBF16 split-precision WMMA (x_hi*w_hi + x_hi*w_lo + x_lo*w_hi) with fp32
// accumulation via v_wmma_f32_16x16x32_bf16. Double-buffered LDS pipeline:
//   iter k: [issue global loads k+1] -> [24 WMMAs from buf k&1]
//           -> [convert+store to buf (k+1)&1] -> one barrier.
// ---------------------------------------------------------------------------

typedef __attribute__((ext_vector_type(16))) __bf16 v16bf;
typedef __attribute__((ext_vector_type(8)))  __bf16 v8bf;
typedef __attribute__((ext_vector_type(4)))  __bf16 v4bf;
typedef __attribute__((ext_vector_type(8)))  float  v8f;
typedef __attribute__((ext_vector_type(4)))  float  v4f;

#define BM 128          // block tile M
#define BN 128          // block tile N
#define BK 32           // K per WMMA step (bf16 K=32)
#define LDT (BK + 8)    // LDS row stride in bf16 (80 B = 20 banks, no conflicts)

__device__ __forceinline__ void split_bf16(float f, __bf16& h, __bf16& l) {
    h = (__bf16)f;              // high part (RNE)
    l = (__bf16)(f - (float)h); // residual -> low part
}

__device__ __forceinline__ v16bf frag16(const __bf16* p0, const __bf16* p1) {
    v8bf c0 = *(const v8bf*)p0;
    v8bf c1 = *(const v8bf*)p1;
    return __builtin_shufflevector(c0, c1,
        0, 1, 2, 3, 4, 5, 6, 7, 8, 9, 10, 11, 12, 13, 14, 15);
}

__global__ __launch_bounds__(256)
void UniCoreLinearFP4_gemm_kernel(const float* __restrict__ X,
                                  const float* __restrict__ W,
                                  const float* __restrict__ S,
                                  const float* __restrict__ Bv,
                                  float* __restrict__ Out,
                                  int M, int N, int K) {
    // Double-buffered hi/lo bf16 tiles: 2 x (4 x 10 KB) = 80 KB LDS.
    __shared__ __bf16 As_hi[2][BM * LDT];
    __shared__ __bf16 As_lo[2][BM * LDT];
    __shared__ __bf16 Bs_hi[2][BN * LDT];
    __shared__ __bf16 Bs_lo[2][BN * LDT];

    const int tid  = threadIdx.x;
    const int lane = tid & 31;
    const int wave = tid >> 5;      // 0..7
    const int wm   = wave >> 2;     // wave row (0..1) -> 64 rows of C
    const int wn   = wave & 3;      // wave col (0..3) -> 32 cols of C
    const int half = lane >> 4;
    const int l16  = lane & 15;

    const int mBlock = blockIdx.y * BM;
    const int nBlock = blockIdx.x * BN;

    // ---- hoisted per-thread staging state (4 x float4 per matrix) ------
    const int quad = tid & 7;       // float4 slot within a 32-float row
    const int row0 = tid >> 3;      // 0..31, stepped by 32
    const float* xp[4];
    const float* wp[4];
    float sc[4];
    int rloc[4];
#pragma unroll
    for (int it = 0; it < 4; ++it) {
        const int r  = row0 + it * 32;
        rloc[it] = r;
        const int gm = min(mBlock + r, M - 1);
        const int gn = min(nBlock + r, N - 1);
        xp[it] = X + (size_t)gm * K + quad * 4;
        wp[it] = W + (size_t)gn * K + quad * 4;
        sc[it] = S[gn];
    }

    v8f acc[4][2];
#pragma unroll
    for (int i = 0; i < 4; ++i)
#pragma unroll
        for (int j = 0; j < 2; ++j)
#pragma unroll
            for (int r = 0; r < 8; ++r) acc[i][j][r] = 0.0f;

    v4f xv[4], wv[4];

    // ---- prologue: load + stage tile 0 into buffer 0 -------------------
#pragma unroll
    for (int it = 0; it < 4; ++it) {
        xv[it] = *(const v4f*)(xp[it]);
        wv[it] = *(const v4f*)(wp[it]);
    }
#pragma unroll
    for (int it = 0; it < 4; ++it) {
        const int off = rloc[it] * LDT + quad * 4;
        v4bf xh, xl, wh, wl;
#pragma unroll
        for (int e = 0; e < 4; ++e) {
            __bf16 h, l;
            split_bf16(xv[it][e], h, l);
            xh[e] = h; xl[e] = l;
            split_bf16(wv[it][e] * sc[it], h, l);
            wh[e] = h; wl[e] = l;
        }
        *(v4bf*)(&As_hi[0][off]) = xh;
        *(v4bf*)(&As_lo[0][off]) = xl;
        *(v4bf*)(&Bs_hi[0][off]) = wh;
        *(v4bf*)(&Bs_lo[0][off]) = wl;
    }
    __syncthreads();

    const int kTiles = K / BK;
    for (int kt = 0; kt < kTiles; ++kt) {
        const int cur = kt & 1;
        const int nxt = cur ^ 1;
        const bool more = (kt + 1) < kTiles;

        // ---- issue next tile's global loads (latency hidden by WMMAs) --
        if (more) {
            const int k0 = (kt + 1) * BK;
#pragma unroll
            for (int it = 0; it < 4; ++it) {
                xv[it] = *(const v4f*)(xp[it] + k0);
                wv[it] = *(const v4f*)(wp[it] + k0);
            }
        }

        // ---- fragments per CDNA5 WMMA VGPR layouts ---------------------
        // A (16x32): lane half=0 -> k 0..7 & 16..23; half=1 -> k 8..15 & 24..31.
        v16bf a_hi[4], a_lo[4];
#pragma unroll
        for (int i = 0; i < 4; ++i) {
            const int m = wm * 64 + i * 16 + l16;
            const __bf16* ph = &As_hi[cur][m * LDT];
            const __bf16* pl = &As_lo[cur][m * LDT];
            a_hi[i] = frag16(ph + half * 8, ph + 16 + half * 8);
            a_lo[i] = frag16(pl + half * 8, pl + 16 + half * 8);
        }
        // B (32x16): lane holds col n = l16, half=0 -> k 0..7 & 16..23; half=1 -> k 8..15 & 24..31.
        v16bf b_hi[2], b_lo[2];
#pragma unroll
        for (int j = 0; j < 2; ++j) {
            const int n = wn * 32 + j * 16 + l16;
            const __bf16* ph = &Bs_hi[cur][n * LDT];
            const __bf16* pl = &Bs_lo[cur][n * LDT];
            b_hi[j] = frag16(ph + half * 8, ph + 16 + half * 8);
            b_lo[j] = frag16(pl + half * 8, pl + 16 + half * 8);
        }

        // ---- 3xBF16 split-precision MMA (fp32-accurate) ----------------
#pragma unroll
        for (int i = 0; i < 4; ++i) {
#pragma unroll
            for (int j = 0; j < 2; ++j) {
                acc[i][j] = __builtin_amdgcn_wmma_f32_16x16x32_bf16(
                    false, a_hi[i], false, b_hi[j], (short)0, acc[i][j], false, false);
                acc[i][j] = __builtin_amdgcn_wmma_f32_16x16x32_bf16(
                    false, a_hi[i], false, b_lo[j], (short)0, acc[i][j], false, false);
                acc[i][j] = __builtin_amdgcn_wmma_f32_16x16x32_bf16(
                    false, a_lo[i], false, b_hi[j], (short)0, acc[i][j], false, false);
            }
        }

        // ---- convert + stage next tile into the other buffer -----------
        if (more) {
#pragma unroll
            for (int it = 0; it < 4; ++it) {
                const int off = rloc[it] * LDT + quad * 4;
                v4bf xh, xl, wh, wl;
#pragma unroll
                for (int e = 0; e < 4; ++e) {
                    __bf16 h, l;
                    split_bf16(xv[it][e], h, l);
                    xh[e] = h; xl[e] = l;
                    split_bf16(wv[it][e] * sc[it], h, l);
                    wh[e] = h; wl[e] = l;
                }
                *(v4bf*)(&As_hi[nxt][off]) = xh;
                *(v4bf*)(&As_lo[nxt][off]) = xl;
                *(v4bf*)(&Bs_hi[nxt][off]) = wh;
                *(v4bf*)(&Bs_lo[nxt][off]) = wl;
            }
        }
        __syncthreads();
    }

    // ---- epilogue: C layout -> row = base + 8*half + r, col = base + l16
#pragma unroll
    for (int j = 0; j < 2; ++j) {
        const int col = nBlock + wn * 32 + j * 16 + l16;
        if (col < N) {
            const float b = Bv[col];
#pragma unroll
            for (int i = 0; i < 4; ++i) {
#pragma unroll
                for (int r = 0; r < 8; ++r) {
                    const int row = mBlock + wm * 64 + i * 16 + half * 8 + r;
                    if (row < M)
                        Out[(size_t)row * N + col] = acc[i][j][r] + b;
                }
            }
        }
    }
}

extern "C" void kernel_launch(void* const* d_in, const int* in_sizes, int n_in,
                              void* d_out, int out_size, void* d_ws, size_t ws_size,
                              hipStream_t stream) {
    const float* X  = (const float*)d_in[0];   // [B,S,K]
    const float* W  = (const float*)d_in[1];   // [N,K]
    const float* S  = (const float*)d_in[2];   // [N,1]
    const float* Bv = (const float*)d_in[3];   // [1,N]
    float* Out = (float*)d_out;

    const int N = in_sizes[2];
    const int K = in_sizes[1] / N;
    const int M = in_sizes[0] / K;

    dim3 grid((N + BN - 1) / BN, (M + BM - 1) / BM);
    dim3 block(256);
    UniCoreLinearFP4_gemm_kernel<<<grid, block, 0, stream>>>(X, W, S, Bv, Out,
                                                            M, N, K);
}
